// BertSelfAttention_377957122480
// MI455X (gfx1250) — matrix operand order repaired and checked by reading, not compiled, not run on hardware
//
#include <hip/hip_runtime.h>
#include <hip/hip_bf16.h>
#include <math.h>

// Problem constants (BERT self-attention)
#define Bn  2
#define Sn  4096
#define Hn  768
#define NHn 12
#define HDn 64

typedef __attribute__((ext_vector_type(16))) __bf16 v16bf;
typedef __attribute__((ext_vector_type(8)))  float  v8f;

union FragAB { uint4 u[2]; v16bf v; };  // 32 bytes: one bf16 A or B fragment per lane

static __device__ __forceinline__ unsigned short f2bf(float f) {
  union { __bf16 b; unsigned short u; } c;
  c.b = (__bf16)f;
  return c.u;
}

static __device__ __forceinline__ FragAB ld_frag(const unsigned short* p, int second_off) {
  FragAB f;
  f.u[0] = *(const uint4*)p;
  f.u[1] = *(const uint4*)(p + second_off);
  return f;
}

// CDNA5 async global->LDS copy (per-lane 16B), tracked by ASYNCcnt.
static __device__ __forceinline__ void async_g2l_b128(unsigned lds_off,
                                                      const unsigned short* g) {
  asm volatile("global_load_async_to_lds_b128 %0, %1, off"
               :: "v"(lds_off), "v"(g) : "memory");
}
static __device__ __forceinline__ void wait_async0() {
  asm volatile("s_wait_asynccnt 0x0" ::: "memory");
}

// ---------------------------------------------------------------------------
// fp32 -> bf16 elementwise conversion
// ---------------------------------------------------------------------------
__global__ void cvt_f32_bf16(const float* __restrict__ in,
                             unsigned short* __restrict__ out, int n) {
  int i = blockIdx.x * blockDim.x + threadIdx.x;
  int stride = gridDim.x * blockDim.x;
  for (; i < n; i += stride) out[i] = f2bf(in[i]);
}

// ---------------------------------------------------------------------------
// QKV projection GEMM: C[M=8192, N=768] = X[8192,768] @ W[768,768]^T + bias
// NT layout -> direct fragment loads. 256 threads = 8 waves; each wave does a
// 32x64 tile (8 WMMAs / K-step, B frags reused across 2 A tiles), with
// explicit next-K-step register double buffering to overlap VMEM with WMMA.
// mode 0: store [B][NH][S][HD] (Q,K); mode 1: store [B][NH][HD][S] (V^T).
// ---------------------------------------------------------------------------
__global__ __launch_bounds__(256) void qkv_gemm(
    const unsigned short* __restrict__ X,
    const unsigned short* __restrict__ W,
    const float* __restrict__ bias,
    unsigned short* __restrict__ out,
    int mode) {
  const int lane  = threadIdx.x & 31;
  const int wave  = threadIdx.x >> 5;
  const int l16   = lane & 15;
  const int hi    = lane >> 4;
  const int mbase = blockIdx.x * 256 + wave * 32;
  const int nbase = blockIdx.y * 64;

  v8f acc[2][4] = {};

  const unsigned short* xrow[2];
  xrow[0] = X + (size_t)(mbase + l16) * Hn + (hi ? 8 : 0);
  xrow[1] = xrow[0] + (size_t)16 * Hn;
  const unsigned short* wr[4];
#pragma unroll
  for (int nt = 0; nt < 4; ++nt)
    wr[nt] = W + (size_t)(nbase + nt * 16 + l16) * Hn + hi * 8;

  FragAB a[2], b[4];
#pragma unroll
  for (int i = 0; i < 2; ++i) a[i] = ld_frag(xrow[i], 16);
#pragma unroll
  for (int nt = 0; nt < 4; ++nt) b[nt] = ld_frag(wr[nt], 16);

  for (int kb = 0; kb < Hn; kb += 32) {
    const int kn = kb + 32;
    FragAB an[2], bn[4];
    if (kn < Hn) {  // preload next K-step while current WMMAs run
#pragma unroll
      for (int i = 0; i < 2; ++i) an[i] = ld_frag(xrow[i] + kn, 16);
#pragma unroll
      for (int nt = 0; nt < 4; ++nt) bn[nt] = ld_frag(wr[nt] + kn, 16);
    }
#pragma unroll
    for (int i = 0; i < 2; ++i)
#pragma unroll
      for (int nt = 0; nt < 4; ++nt)
        acc[i][nt] = __builtin_amdgcn_wmma_f32_16x16x32_bf16(
            false, a[i].v, false, b[nt].v, (short)0, acc[i][nt], false, false);
    if (kn < Hn) {
#pragma unroll
      for (int i = 0; i < 2; ++i) a[i] = an[i];
#pragma unroll
      for (int nt = 0; nt < 4; ++nt) b[nt] = bn[nt];
    }
  }

#pragma unroll
  for (int nt = 0; nt < 4; ++nt) {
    const int col = nbase + nt * 16 + l16;
    const float bv = bias[col];
    const int h = col >> 6, d = col & 63;
#pragma unroll
    for (int i = 0; i < 2; ++i) {
#pragma unroll
      for (int r = 0; r < 8; ++r) {
        const int g = mbase + i * 16 + r + hi * 8;  // C layout: M = r / r+8
        const int bb = g >> 12, s = g & 4095;
        const unsigned short val = f2bf(acc[i][nt][r] + bv);
        size_t idx;
        if (mode == 0)
          idx = (((size_t)(bb * NHn + h) * Sn + s) * HDn + d);
        else
          idx = (((size_t)(bb * NHn + h) * HDn + d) * Sn + s);
        out[idx] = val;
      }
    }
  }
}

// ---------------------------------------------------------------------------
// Flash attention: grid (S/64, NH, B), 128 threads = 4 waves.
// Each wave owns 16 query rows. K/V tiles (32 keys) are double-buffered in
// LDS and filled with async global->LDS copies (ASYNCcnt), so the next tile
// streams in while the current tile's WMMAs + softmax execute.
// ---------------------------------------------------------------------------
__global__ __launch_bounds__(128) void attn(
    const unsigned short* __restrict__ Q,   // [B][NH][S][HD]
    const unsigned short* __restrict__ K,   // [B][NH][S][HD]
    const unsigned short* __restrict__ Vt,  // [B][NH][HD][S]
    const float* __restrict__ mask,         // [B][S] additive
    float* __restrict__ out) {              // [B][S][H]
  __shared__ __align__(16) unsigned short Kt[2][32 * 64];  // 32 keys x 64 hd
  __shared__ __align__(16) unsigned short Vl[2][64 * 32];  // 64 hd x 32 keys
  __shared__ __align__(16) unsigned short Pb[4][16 * 32];  // per-wave P staging

  const int lane = threadIdx.x & 31;
  const int wave = threadIdx.x >> 5;
  const int l16  = lane & 15;
  const int hi   = lane >> 4;
  const int b    = blockIdx.z;
  const int h    = blockIdx.y;
  const int q0   = blockIdx.x * 64 + wave * 16;

  const size_t head = (size_t)(b * NHn + h) * Sn * HDn;
  const unsigned short* Qh = Q + head;
  const unsigned short* Kh = K + head;
  const unsigned short* Vh = Vt + head;
  const float* mrow = mask + (size_t)b * Sn;

  // Q A-fragments (resident for whole loop): kk = 0 (hd 0..31), 1 (hd 32..63)
  FragAB aq[2];
  {
    const unsigned short* qrow = Qh + (size_t)(q0 + l16) * HDn + (hi ? 8 : 0);
#pragma unroll
    for (int kk = 0; kk < 2; ++kk) aq[kk] = ld_frag(qrow + kk * 32, 16);
  }

  v8f O[4] = {};
  float mrun[8], lrun[8];
#pragma unroll
  for (int r = 0; r < 8; ++r) { mrun[r] = -__builtin_inff(); lrun[r] = 0.0f; }

  const float sc = 0.125f;  // 1/sqrt(64)

  // Issue a 32-key K/V tile into LDS buffer `bf` with async copies (4 wave ops).
  auto issue_tile = [&](int kb, int bf) {
    const unsigned kbase = (unsigned)(size_t)&Kt[bf][0];
    const unsigned vbase = (unsigned)(size_t)&Vl[bf][0];
    const unsigned short* gk = Kh + (size_t)kb * HDn;
#pragma unroll
    for (int i = 0; i < 2; ++i) {
      const int c = threadIdx.x + i * 128;           // 256 x 16B chunks
      async_g2l_b128(kbase + (unsigned)c * 16, gk + c * 8);
      const int d = c >> 2, part = c & 3;            // Vt rows: 4 chunks each
      async_g2l_b128(vbase + (unsigned)c * 16,
                     Vh + (size_t)d * Sn + kb + part * 8);
    }
  };

  issue_tile(0, 0);
  int buf = 0;

  for (int kb = 0; kb < Sn; kb += 32) {
    wait_async0();        // current tile's async writes (issued last iter) done
    __syncthreads();      // tile visible to all waves; prev reads of buf^1 done
    if (kb + 32 < Sn) issue_tile(kb + 32, buf ^ 1);  // overlap next tile

    // scores: 16 q x 32 keys = 4 WMMAs (hd-dim 64 -> 2 per 16-col tile)
    v8f s[2] = {};
#pragma unroll
    for (int nt = 0; nt < 2; ++nt) {
#pragma unroll
      for (int kk = 0; kk < 2; ++kk) {
        const unsigned short* p = &Kt[buf][(nt * 16 + l16) * 64 + kk * 32 + hi * 8];
        FragAB bk = ld_frag(p, 16);
        s[nt] = __builtin_amdgcn_wmma_f32_16x16x32_bf16(
            false, aq[kk].v, false, bk.v, (short)0, s[nt], false, false);
      }
    }

    // online softmax (rows live across 16 lanes of each half-wave)
    const float mk0 = mrow[kb + l16];
    const float mk1 = mrow[kb + 16 + l16];
#pragma unroll
    for (int r = 0; r < 8; ++r) {
      float a0 = s[0][r] * sc + mk0;
      float a1 = s[1][r] * sc + mk1;
      float mx = fmaxf(a0, a1);
#pragma unroll
      for (int off = 8; off; off >>= 1) mx = fmaxf(mx, __shfl_xor(mx, off, 16));
      const float mnew = fmaxf(mrun[r], mx);
      const float corr = __expf(mrun[r] - mnew);
      mrun[r] = mnew;
      lrun[r] *= corr;
      O[0][r] *= corr; O[1][r] *= corr; O[2][r] *= corr; O[3][r] *= corr;
      const float p0 = __expf(a0 - mnew);
      const float p1 = __expf(a1 - mnew);
      float rs = p0 + p1;
#pragma unroll
      for (int off = 8; off; off >>= 1) rs += __shfl_xor(rs, off, 16);
      lrun[r] += rs;
      // stage P (bf16) into per-wave LDS, row-major 16x32
      unsigned short* pb = &Pb[wave][(r + hi * 8) * 32];
      pb[l16]      = f2bf(p0);
      pb[16 + l16] = f2bf(p1);
    }

    // P (16x32) A-fragment from LDS; V B-fragments contiguous thanks to V^T
    FragAB ap = ld_frag(&Pb[wave][l16 * 32 + (hi ? 8 : 0)], 16);
#pragma unroll
    for (int ot = 0; ot < 4; ++ot) {
      const unsigned short* p = &Vl[buf][(ot * 16 + l16) * 32 + hi * 8];
      FragAB bv = ld_frag(p, 16);
      O[ot] = __builtin_amdgcn_wmma_f32_16x16x32_bf16(
          false, ap.v, false, bv.v, (short)0, O[ot], false, false);
    }
    buf ^= 1;
  }

  // epilogue: normalize by row sum, write fp32 ctx [B][S][H]
#pragma unroll
  for (int ot = 0; ot < 4; ++ot) {
#pragma unroll
    for (int r = 0; r < 8; ++r) {
      const int s = q0 + r + hi * 8;
      const float val = O[ot][r] / lrun[r];
      out[((size_t)(b * Sn + s)) * Hn + h * HDn + ot * 16 + l16] = val;
    }
  }
}

// ---------------------------------------------------------------------------
extern "C" void kernel_launch(void* const* d_in, const int* in_sizes, int n_in,
                              void* d_out, int out_size, void* d_ws, size_t ws_size,
                              hipStream_t stream) {
  const float* hs   = (const float*)d_in[0];
  const float* mask = (const float*)d_in[1];
  const float* Wq   = (const float*)d_in[2];
  const float* bq   = (const float*)d_in[3];
  const float* Wk   = (const float*)d_in[4];
  const float* bk   = (const float*)d_in[5];
  const float* Wv   = (const float*)d_in[6];
  const float* bv   = (const float*)d_in[7];
  float* out = (float*)d_out;

  const int nX = Bn * Sn * Hn;      // 6,291,456
  const int nW = Hn * Hn;           //   589,824

  unsigned short* Xbf = (unsigned short*)d_ws;
  unsigned short* Wqb = Xbf + nX;
  unsigned short* Wkb = Wqb + nW;
  unsigned short* Wvb = Wkb + nW;
  unsigned short* Qb  = Wvb + nW;
  unsigned short* Kb  = Qb + nX;
  unsigned short* Vtb = Kb + nX;    // total ~54 MB of ws

  cvt_f32_bf16<<<2048, 256, 0, stream>>>(hs, Xbf, nX);
  cvt_f32_bf16<<<512, 256, 0, stream>>>(Wq, Wqb, nW);
  cvt_f32_bf16<<<512, 256, 0, stream>>>(Wk, Wkb, nW);
  cvt_f32_bf16<<<512, 256, 0, stream>>>(Wv, Wvb, nW);

  dim3 gg(32, 12, 1);  // M/256, N/64
  qkv_gemm<<<gg, 256, 0, stream>>>(Xbf, Wqb, bq, Qb, 0);
  qkv_gemm<<<gg, 256, 0, stream>>>(Xbf, Wkb, bk, Kb, 0);
  qkv_gemm<<<gg, 256, 0, stream>>>(Xbf, Wvb, bv, Vtb, 1);

  dim3 ga(Sn / 64, NHn, Bn);
  attn<<<ga, 128, 0, stream>>>(Qb, Kb, Vtb, mask, out);
}
